// Model_61108794688040
// MI455X (gfx1250) — matrix operand order repaired and checked by reading, not compiled, not run on hardware
//
#include <hip/hip_runtime.h>
#include <hip/hip_bf16.h>

typedef _Float16 half_t;
typedef half_t v16h __attribute__((ext_vector_type(16)));
typedef half_t v8h  __attribute__((ext_vector_type(8)));
typedef float  v8f  __attribute__((ext_vector_type(8)));

// ---------------------------------------------------------------------------
// Conv tower (tiny fraction of total FLOPs): direct conv + relu, batch-stat BN
// ---------------------------------------------------------------------------
__global__ void conv_relu(const float* __restrict__ x, const float* __restrict__ w,
                          const float* __restrict__ bias, float* __restrict__ y,
                          int Cin, int Cout, int Hin, int Hout) {
  int idx = blockIdx.x * 256 + threadIdx.x;
  int total = 64 * Cout * Hout * Hout;
  if (idx >= total) return;
  int wo = idx % Hout; int t = idx / Hout;
  int ho = t % Hout;   t /= Hout;
  int co = t % Cout;   int b = t / Cout;
  float acc = bias[co];
  for (int ci = 0; ci < Cin; ++ci) {
    for (int kh = 0; kh < 3; ++kh) {
      int hi = ho * 2 - 1 + kh;
      if (hi < 0 || hi >= Hin) continue;
      for (int kw = 0; kw < 3; ++kw) {
        int wi = wo * 2 - 1 + kw;
        if (wi < 0 || wi >= Hin) continue;
        acc += x[((size_t)(b * Cin + ci) * Hin + hi) * Hin + wi] *
               w[((co * Cin + ci) * 3 + kh) * 3 + kw];
      }
    }
  }
  y[idx] = acc > 0.f ? acc : 0.f;
}

__global__ void bn_stats(const float* __restrict__ x, float* __restrict__ mv,
                         int C, int HW) {
  int c = blockIdx.x;
  int n = 64 * HW;
  float s = 0.f, sq = 0.f;
  for (int i = threadIdx.x; i < n; i += 256) {
    int b = i / HW, p = i % HW;
    float v = x[((size_t)(b * C + c)) * HW + p];
    s += v; sq += v * v;
  }
  __shared__ float ss[256], s2[256];
  ss[threadIdx.x] = s; s2[threadIdx.x] = sq;
  __syncthreads();
  for (int o = 128; o > 0; o >>= 1) {
    if (threadIdx.x < o) {
      ss[threadIdx.x] += ss[threadIdx.x + o];
      s2[threadIdx.x] += s2[threadIdx.x + o];
    }
    __syncthreads();
  }
  if (threadIdx.x == 0) {
    float m = ss[0] / (float)n;
    float var = s2[0] / (float)n - m * m;
    mv[c * 2] = m;
    mv[c * 2 + 1] = rsqrtf(var + 1e-5f);
  }
}

__global__ void bn_apply(float* __restrict__ x, const float* __restrict__ mv,
                         const float* __restrict__ g, const float* __restrict__ bb,
                         int C, int HW) {
  int idx = blockIdx.x * 256 + threadIdx.x;
  int total = 64 * C * HW;
  if (idx >= total) return;
  int c = (idx / HW) % C;
  x[idx] = g[c] * (x[idx] - mv[c * 2]) * mv[c * 2 + 1] + bb[c];
}

// ---------------------------------------------------------------------------
// Layer-1 decomposition: A[b,i] = o_i @ gw1[0:26], B[b,j] = o_j @ gw1[26:52]
// ---------------------------------------------------------------------------
__global__ void features(const float* __restrict__ x4, const float* __restrict__ gw1,
                         float* __restrict__ Afp, float* __restrict__ Bfp) {
  int bi = blockIdx.x;            // b*64 + i
  int b = bi >> 6, i = bi & 63;
  __shared__ float o[26];
  if (threadIdx.x < 24)       o[threadIdx.x] = x4[((size_t)(b * 24 + threadIdx.x) * 8 + (i >> 3)) * 8 + (i & 7)];
  else if (threadIdx.x == 24) o[24] = (float)(i >> 3) * 0.125f;
  else if (threadIdx.x == 25) o[25] = (float)(i & 7) * 0.125f;
  __syncthreads();
  int c = threadIdx.x;
  float a = 0.f, bv = 0.f;
  for (int f = 0; f < 26; ++f) {
    a  += o[f] * gw1[f * 256 + c];
    bv += o[f] * gw1[(26 + f) * 256 + c];
  }
  Afp[(size_t)bi * 256 + c] = a;
  Bfp[(size_t)bi * 256 + c] = bv;
}

__global__ void cq_kernel(const float* __restrict__ q, const float* __restrict__ gw1,
                          const float* __restrict__ gb1, float* __restrict__ Cq) {
  int b = blockIdx.x, c = threadIdx.x;
  float s = gb1[c];
  for (int f = 0; f < 11; ++f) s += q[b * 11 + f] * gw1[(52 + f) * 256 + c];
  Cq[b * 256 + c] = s;
}

// ---------------------------------------------------------------------------
// Repack gw2/gw3/gw4 (256x256 fp32, [fin][fout]) into WMMA B-fragment order,
// f16. contiguous k order used here (A frags match, not the ISA table order): lane -> N=l&15, K half = (l>>4)*16,
// element e -> K = (l>>4)*16 + e within k-block.
// Storage: [kb(8)][nt(16)][lane(32)][e(16)] halfs -> aligned 32B per lane.
// ---------------------------------------------------------------------------
__global__ void repack_w(const float* __restrict__ w2, const float* __restrict__ w3,
                         const float* __restrict__ w4, half_t* __restrict__ wfrag) {
  int idx = blockIdx.x * 256 + threadIdx.x;   // 3 * 65536 total
  int layer = idx >> 16;
  int rem = idx & 65535;
  int e    = rem & 15;
  int lane = (rem >> 4) & 31;
  int nt   = (rem >> 9) & 15;
  int kb   = rem >> 13;
  int k = kb * 32 + ((lane >> 4) << 4) + e;
  int n = nt * 16 + (lane & 15);
  const float* w = (layer == 0) ? w2 : ((layer == 1) ? w3 : w4);
  wfrag[idx] = (half_t)w[k * 256 + n];
}

// ---------------------------------------------------------------------------
// Fused relation-network core. WG = 256 thr (8 wave32), owns 256 pair-rows of
// one batch; each wave runs TWO 16x256 tiles sharing B fragments -> two
// independent v_wmma_f32_16x16x32_f16 chains per n-tile. B fragments are
// software double-buffered so ds_loads overlap the WMMA chain; accumulators
// zero-init (inline SRC2=0 on first WMMA) with bias folded into the epilogue.
// ---------------------------------------------------------------------------
extern __shared__ char smem_raw[];

__global__ __launch_bounds__(256)
void rn_main(const float* __restrict__ Afp, const float* __restrict__ Bfp,
             const float* __restrict__ Cq,  const half_t* __restrict__ wfrag,
             const float* __restrict__ gb2, const float* __restrict__ gb3,
             const float* __restrict__ gb4, float* __restrict__ partial) {
  half_t* wbuf   = (half_t*)smem_raw;                       // 65536 halfs = 128KB
  half_t* stage  = (half_t*)(smem_raw + 131072);            // 8 waves * 32x256 halfs
  float*  sumbuf = (float*)(smem_raw + 131072 + 131072);    // 256 floats

  int wg = blockIdx.x;
  int b = wg >> 4;
  int rowbase = (wg & 15) * 256;
  int tid = threadIdx.x;
  int wave = tid >> 5, lane = tid & 31;
  half_t* st = stage + wave * 8192;                         // 32 rows x 256

  sumbuf[tid] = 0.f;

  // ---- layer 1: h1 = relu(A_i + B_j + Cq), written as f16 tile -------------
  const float* Cb = Cq + (size_t)b * 256;
  for (int r = 0; r < 32; ++r) {
    int m = rowbase + wave * 32 + r;                 // pair index within batch
    const float* Ai = Afp + ((size_t)b * 64 + (m >> 6)) * 256;
    const float* Bj = Bfp + ((size_t)b * 64 + (m & 63)) * 256;
    for (int c = lane; c < 256; c += 32) {
      float v = Ai[c] + Bj[c] + Cb[c];
      st[r * 256 + c] = (half_t)(v > 0.f ? v : 0.f);
    }
  }

  // ---- layers 2..4: dual-tile, double-buffered WMMA pipeline ---------------
  int mrow = lane & 15;
  int koff = (lane >> 4) * 16;           // A-frag: lanes 16-31 start 16 halfs in

  auto loadB = [&](half_t* wb, int kb, int nt) -> v16h {
    int boff = ((kb * 16 + nt) * 32 + lane) * 16;
    v8h b0 = *(const v8h*)(wb + boff);
    v8h b1 = *(const v8h*)(wb + boff + 8);
    return __builtin_shufflevector(b0, b1, 0, 1, 2, 3, 4, 5, 6, 7,
                                   8, 9, 10, 11, 12, 13, 14, 15);
  };

  for (int L = 0; L < 3; ++L) {
    __syncthreads();
    // stage fragment-ordered weights into LDS (b128 copies)
    {
      const uint4* src = (const uint4*)(wfrag + (size_t)L * 65536);
      uint4* dst = (uint4*)wbuf;
      for (int kidx = tid; kidx < 8192; kidx += 256) dst[kidx] = src[kidx];
    }
    __syncthreads();
    // prefetch next layer's weights into L2 while we compute this one
    if (L < 2) {
      const char* nxt = (const char*)(wfrag + (size_t)(L + 1) * 65536);
      for (int pf = tid; pf < 1024; pf += 256)
        __builtin_prefetch(nxt + (size_t)pf * 128, 0, 1);
    }

    // load A fragments for both tiles (two contiguous 8-half runs each)
    v16h a0[8], a1[8];
#pragma unroll
    for (int kb = 0; kb < 8; ++kb) {
      v8h lo0 = *(const v8h*)(st + mrow * 256 + kb * 32 + koff);
      v8h hi0 = *(const v8h*)(st + mrow * 256 + kb * 32 + koff + 8);
      a0[kb] = __builtin_shufflevector(lo0, hi0, 0, 1, 2, 3, 4, 5, 6, 7,
                                       8, 9, 10, 11, 12, 13, 14, 15);
      v8h lo1 = *(const v8h*)(st + 4096 + mrow * 256 + kb * 32 + koff);
      v8h hi1 = *(const v8h*)(st + 4096 + mrow * 256 + kb * 32 + koff + 8);
      a1[kb] = __builtin_shufflevector(lo1, hi1, 0, 1, 2, 3, 4, 5, 6, 7,
                                       8, 9, 10, 11, 12, 13, 14, 15);
    }

    const float* gb = (L == 0) ? gb2 : ((L == 1) ? gb3 : gb4);
    float biasv[16];
#pragma unroll
    for (int nt = 0; nt < 16; ++nt) biasv[nt] = gb[nt * 16 + (lane & 15)];

    for (int nt = 0; nt < 16; ++nt) {
      v8f acc0 = {};                      // inline SRC2=0 on first WMMA
      v8f acc1 = {};
      v16h bf = loadB(wbuf, 0, nt);       // software pipeline: preload kb=0
#pragma unroll
      for (int kb = 0; kb < 8; ++kb) {
        v16h bfn = bf;
        if (kb < 7) bfn = loadB(wbuf, kb + 1, nt);   // overlaps WMMA below
        acc0 = __builtin_amdgcn_wmma_f32_16x16x32_f16(
            false, a0[kb], false, bf, (short)0, acc0, false, false);
        acc1 = __builtin_amdgcn_wmma_f32_16x16x32_f16(
            false, a1[kb], false, bf, (short)0, acc1, false, false);
        bf = bfn;
      }
      float bias = biasv[nt];
      if (L < 2) {
        // bias + relu + restage as next layer's input (C/D layout -> row/col)
#pragma unroll
        for (int r = 0; r < 8; ++r) {
          float v0 = acc0[r] + bias; v0 = v0 > 0.f ? v0 : 0.f;
          float v1 = acc1[r] + bias; v1 = v1 > 0.f ? v1 : 0.f;
          int row = r + ((lane >> 4) << 3);
          int col = nt * 16 + (lane & 15);
          st[row * 256 + col]        = (half_t)v0;
          st[(16 + row) * 256 + col] = (half_t)v1;
        }
      } else {
        // h4: bias + relu + sum over the 32 rows of this wave's tiles
        float s = 0.f;
#pragma unroll
        for (int r = 0; r < 8; ++r) {
          float v0 = acc0[r] + bias;
          float v1 = acc1[r] + bias;
          s += (v0 > 0.f ? v0 : 0.f);
          s += (v1 > 0.f ? v1 : 0.f);
        }
        atomicAdd(&sumbuf[nt * 16 + (lane & 15)], s);   // ds_add_f32
      }
    }
  }
  __syncthreads();
  partial[(size_t)wg * 256 + tid] = sumbuf[tid];
}

// ---------------------------------------------------------------------------
// Final head: mean over 4096 pairs (16 partials/batch), f_phi MLP, logits.
// ---------------------------------------------------------------------------
__global__ void fphi(const float* __restrict__ partial,
                     const float* __restrict__ fw1, const float* __restrict__ fb1,
                     const float* __restrict__ fw2, const float* __restrict__ fb2,
                     const float* __restrict__ fw3, const float* __restrict__ fb3,
                     float* __restrict__ out) {
  int b = blockIdx.x, c = threadIdx.x;
  __shared__ float g[256], h[256];
  float s = 0.f;
  for (int w = 0; w < 16; ++w) s += partial[((size_t)(b * 16 + w)) * 256 + c];
  g[c] = s * (1.f / 4096.f);
  __syncthreads();
  float a = fb1[c];
  for (int k = 0; k < 256; ++k) a += g[k] * fw1[k * 256 + c];
  h[c] = a > 0.f ? a : 0.f;
  __syncthreads();
  float a2 = fb2[c];
  for (int k = 0; k < 256; ++k) a2 += h[k] * fw2[k * 256 + c];
  __syncthreads();
  g[c] = a2 > 0.f ? a2 : 0.f;
  __syncthreads();
  if (c < 10) {
    float a3 = fb3[c];
    for (int k = 0; k < 256; ++k) a3 += g[k] * fw3[k * 10 + c];
    out[b * 10 + c] = a3;
  }
}

// ---------------------------------------------------------------------------
extern "C" void kernel_launch(void* const* d_in, const int* in_sizes, int n_in,
                              void* d_out, int out_size, void* d_ws, size_t ws_size,
                              hipStream_t stream) {
  const float* img = (const float*)d_in[0];
  const float* q   = (const float*)d_in[1];
  const float* cw[4] = {(const float*)d_in[2], (const float*)d_in[6],
                        (const float*)d_in[10], (const float*)d_in[14]};
  const float* cb[4] = {(const float*)d_in[3], (const float*)d_in[7],
                        (const float*)d_in[11], (const float*)d_in[15]};
  const float* bg[4] = {(const float*)d_in[4], (const float*)d_in[8],
                        (const float*)d_in[12], (const float*)d_in[16]};
  const float* bbp[4] = {(const float*)d_in[5], (const float*)d_in[9],
                         (const float*)d_in[13], (const float*)d_in[17]};
  const float* gw1 = (const float*)d_in[18];
  const float* gb1 = (const float*)d_in[19];
  const float* gw2 = (const float*)d_in[20];
  const float* gb2 = (const float*)d_in[21];
  const float* gw3 = (const float*)d_in[22];
  const float* gb3 = (const float*)d_in[23];
  const float* gw4 = (const float*)d_in[24];
  const float* gb4 = (const float*)d_in[25];
  const float* fw1 = (const float*)d_in[26];
  const float* fb1 = (const float*)d_in[27];
  const float* fw2 = (const float*)d_in[28];
  const float* fb2 = (const float*)d_in[29];
  const float* fw3 = (const float*)d_in[30];
  const float* fb3 = (const float*)d_in[31];
  float* out = (float*)d_out;

  // workspace layout (bytes)
  char* base = (char*)d_ws;
  float*  x1      = (float*)(base + 0);           // 64*24*64*64
  float*  x2      = (float*)(base + 25165824);    // 64*24*32*32
  float*  x3      = (float*)(base + 31457280);    // 64*24*16*16
  float*  x4      = (float*)(base + 33030144);    // 64*24*8*8
  float*  mv      = (float*)(base + 33423360);    // 24*2 (+pad)
  float*  Afp     = (float*)(base + 33423616);    // 64*64*256
  float*  Bfp     = (float*)(base + 37617920);    // 64*64*256
  float*  Cq      = (float*)(base + 41812224);    // 64*256
  half_t* wfrag   = (half_t*)(base + 41877760);   // 3*65536 halfs
  float*  partial = (float*)(base + 42270976);    // 1024*256
  if (ws_size < 43319552) return;

  // conv tower
  conv_relu<<<(64 * 24 * 64 * 64) / 256, 256, 0, stream>>>(img, cw[0], cb[0], x1, 3, 24, 128, 64);
  bn_stats <<<24, 256, 0, stream>>>(x1, mv, 24, 64 * 64);
  bn_apply <<<(64 * 24 * 64 * 64) / 256, 256, 0, stream>>>(x1, mv, bg[0], bbp[0], 24, 64 * 64);

  conv_relu<<<(64 * 24 * 32 * 32) / 256, 256, 0, stream>>>(x1, cw[1], cb[1], x2, 24, 24, 64, 32);
  bn_stats <<<24, 256, 0, stream>>>(x2, mv, 24, 32 * 32);
  bn_apply <<<(64 * 24 * 32 * 32) / 256, 256, 0, stream>>>(x2, mv, bg[1], bbp[1], 24, 32 * 32);

  conv_relu<<<(64 * 24 * 16 * 16) / 256, 256, 0, stream>>>(x2, cw[2], cb[2], x3, 24, 24, 32, 16);
  bn_stats <<<24, 256, 0, stream>>>(x3, mv, 24, 16 * 16);
  bn_apply <<<(64 * 24 * 16 * 16) / 256, 256, 0, stream>>>(x3, mv, bg[2], bbp[2], 24, 16 * 16);

  conv_relu<<<(64 * 24 * 8 * 8) / 256, 256, 0, stream>>>(x3, cw[3], cb[3], x4, 24, 24, 16, 8);
  bn_stats <<<24, 256, 0, stream>>>(x4, mv, 24, 8 * 8);
  bn_apply <<<(64 * 24 * 8 * 8) / 256, 256, 0, stream>>>(x4, mv, bg[3], bbp[3], 24, 8 * 8);

  // layer-1 decomposition + weight repack
  features <<<64 * 64, 256, 0, stream>>>(x4, gw1, Afp, Bfp);
  cq_kernel<<<64, 256, 0, stream>>>(q, gw1, gb1, Cq);
  repack_w <<<(3 * 65536) / 256, 256, 0, stream>>>(gw2, gw3, gw4, wfrag);

  // fused WMMA relation core: 64 batches * 16 tiles of 256 pair-rows
  rn_main<<<1024, 256, 131072 + 131072 + 1024, stream>>>(
      Afp, Bfp, Cq, wfrag, gb2, gb3, gb4, partial);

  // head
  fphi<<<64, 256, 0, stream>>>(partial, fw1, fb1, fw2, fb2, fw3, fb3, out);
}
